// FastSelfAttention_1881195676076
// MI455X (gfx1250) — compile-verified, matrix operand order checked by reading, not run on hardware
//
#include <hip/hip_runtime.h>
#include <math.h>

#define BATCH 8
#define SEQ   4096
#define HD    512
#define NHEAD 8
#define DHEAD 64

typedef __bf16 bf16;
typedef __attribute__((ext_vector_type(16))) __bf16 bf16x16;
typedef __attribute__((ext_vector_type(8)))  float  f32x8;

union BFU { bf16x16 v; uint4 q[2]; };

// ---------------------------------------------------------------------------
// B-fragment (32x16 K x N, bf16) for v_wmma_f32_16x16x32_bf16.
// Lane r=lane&15 -> column n0+r ; half=lane>>4 selects K pattern
// elements 0..7  <- K = k0 + 8*half + 0..7
// elements 8..15 <- K = k0 + 16 + 8*half + 0..7
// W is row-major (N x K), so both runs are contiguous 16B loads.
// ---------------------------------------------------------------------------
__device__ __forceinline__ bf16x16 load_b_frag(const bf16* __restrict__ W, int n0,
                                               int r, int k0, int kb0, int kb1) {
  const bf16* p = W + (size_t)(n0 + r) * HD + k0;
  BFU u;
  u.q[0] = *(const uint4*)(p + kb0);
  u.q[1] = *(const uint4*)(p + kb1);
  return u.v;
}

// ---------------------------------------------------------------------------
// A-fragment (16x32 M x K) from fp32 source, converted to bf16 on the fly,
// with compile-time-optional per-feature gate (gate[k], per batch) fused in.
// Lane r -> row row0+r ; same K pattern as B.
// ---------------------------------------------------------------------------
template <bool HAS_GATE>
__device__ __forceinline__ bf16x16 load_a_frag(const float* __restrict__ A,
                                               const float* __restrict__ gate,
                                               int row0, int r, int k0,
                                               int kb0, int kb1) {
  const float* p = A + (size_t)(row0 + r) * HD + k0;
  float t[16];
  *(float4*)(t + 0)  = *(const float4*)(p + kb0);
  *(float4*)(t + 4)  = *(const float4*)(p + kb0 + 4);
  *(float4*)(t + 8)  = *(const float4*)(p + kb1);
  *(float4*)(t + 12) = *(const float4*)(p + kb1 + 4);
  if constexpr (HAS_GATE) {
    const float* g = gate + k0;
    float gt[16];
    *(float4*)(gt + 0)  = *(const float4*)(g + kb0);
    *(float4*)(gt + 4)  = *(const float4*)(g + kb0 + 4);
    *(float4*)(gt + 8)  = *(const float4*)(g + kb1);
    *(float4*)(gt + 12) = *(const float4*)(g + kb1 + 4);
#pragma unroll
    for (int i = 0; i < 16; ++i) t[i] *= gt[i];
  }
  bf16x16 v;
#pragma unroll
  for (int i = 0; i < 16; ++i) v[i] = (bf16)t[i];
  return v;
}

// ---------------------------------------------------------------------------
// C[M,HD] = (A ⊙ gate_b) (M,HD fp32) x W^T (W: HD x HD bf16 row-major)
//           + bias [+ resid]
// Block: 256 threads = 8 waves, arranged 4(M) x 2(N); wave tile 32x64.
// Block tile 128x128. grid = (M/128, HD/128).
// Gate/resid presence is a template parameter -> no exec-mask divergence in
// the k-loop or epilogue.
// ---------------------------------------------------------------------------
template <bool HAS_GATE, bool HAS_RESID>
__global__ __launch_bounds__(256) void gemm_bf16_kernel(
    const float* __restrict__ A, const unsigned short* __restrict__ Wus,
    const float* __restrict__ bias, const float* __restrict__ gate,
    const float* __restrict__ resid, float* __restrict__ C) {
  const bf16* W = (const bf16*)Wus;
  const int lane = threadIdx.x & 31;
  const int wave = threadIdx.x >> 5;
  const int wm = wave >> 1;          // 0..3
  const int wn = wave & 1;           // 0..1
  const int m0 = blockIdx.x * 128 + wm * 32;
  const int n0 = blockIdx.y * 128 + wn * 64;
  const int half = lane >> 4;
  const int r = lane & 15;
  const int kb0 = 8 * half;
  const int kb1 = 16 + 8 * half;

  const float* gb = HAS_GATE ? gate + (size_t)(m0 / SEQ) * HD : (const float*)0;

  f32x8 acc[2][4];
  const f32x8 zero = {0.f, 0.f, 0.f, 0.f, 0.f, 0.f, 0.f, 0.f};
#pragma unroll
  for (int mi = 0; mi < 2; ++mi)
#pragma unroll
    for (int ni = 0; ni < 4; ++ni) acc[mi][ni] = zero;

  for (int k0 = 0; k0 < HD; k0 += 32) {
    bf16x16 a[2];
    a[0] = load_a_frag<HAS_GATE>(A, gb, m0,      r, k0, kb0, kb1);
    a[1] = load_a_frag<HAS_GATE>(A, gb, m0 + 16, r, k0, kb0, kb1);
    bf16x16 b[4];
#pragma unroll
    for (int ni = 0; ni < 4; ++ni)
      b[ni] = load_b_frag(W, n0 + 16 * ni, r, k0, kb0, kb1);

    if (k0 + 32 < HD) {  // speculative prefetch of next A k-slab
      __builtin_prefetch(A + (size_t)(m0 + r) * HD + k0 + 32, 0, 1);
    }

#pragma unroll
    for (int mi = 0; mi < 2; ++mi)
#pragma unroll
      for (int ni = 0; ni < 4; ++ni)
        acc[mi][ni] = __builtin_amdgcn_wmma_f32_16x16x32_bf16(
            false, a[mi], false, b[ni], (short)0, acc[mi][ni], false, false);
  }

  // Epilogue. C/D layout: lane half=0 -> rows m+0..7 (VGPR e), half=1 -> m+8..15.
#pragma unroll
  for (int ni = 0; ni < 4; ++ni) {
    const int col = n0 + ni * 16 + r;
    const float bv = bias[col];
#pragma unroll
    for (int mi = 0; mi < 2; ++mi) {
#pragma unroll
      for (int e = 0; e < 8; ++e) {
        const int row = m0 + mi * 16 + half * 8 + e;
        float v = acc[mi][ni][e] + bv;
        if constexpr (HAS_RESID) v += resid[(size_t)row * HD + col];
        C[(size_t)row * HD + col] = v;
      }
    }
  }
}

// ---------------------------------------------------------------------------
// score[row,h] = (sum_j X[row,j]*(gate?gate[b,j]:1)*Wa[h,j] + ba[h]) * 1/sqrt(64)
// one wave per row; block = 8 waves.
// ---------------------------------------------------------------------------
template <bool HAS_GATE>
__global__ __launch_bounds__(256) void score_kernel(
    const float* __restrict__ X, const float* __restrict__ gate,
    const float* __restrict__ Wa, const float* __restrict__ ba,
    float* __restrict__ score) {
  const int row  = blockIdx.x * 8 + (threadIdx.x >> 5);
  const int lane = threadIdx.x & 31;
  const float* x = X + (size_t)row * HD;
  float xv[16];
#pragma unroll
  for (int i = 0; i < 16; ++i) xv[i] = x[lane + 32 * i];
  if constexpr (HAS_GATE) {
    const float* g = gate + (size_t)(row / SEQ) * HD;
#pragma unroll
    for (int i = 0; i < 16; ++i) xv[i] *= g[lane + 32 * i];
  }
  const float scale = 0.125f;  // 1/sqrt(DHEAD)
#pragma unroll
  for (int h = 0; h < NHEAD; ++h) {
    const float* w = Wa + (size_t)h * HD;
    float d = 0.f;
#pragma unroll
    for (int i = 0; i < 16; ++i) d += xv[i] * w[lane + 32 * i];
#pragma unroll
    for (int off = 16; off > 0; off >>= 1) d += __shfl_down(d, off, 32);
    if (lane == 0) score[(size_t)row * NHEAD + h] = (d + ba[h]) * scale;
  }
}

// ---------------------------------------------------------------------------
// Softmax over S of score[b,:,h], then pooled[b, h*64+d] =
//   (mult?mult[b,h*64+d]:1) * sum_s softmax_w[s] * X[b,s,h*64+d]
// one block per (b,h); 256 threads.
// ---------------------------------------------------------------------------
template <bool HAS_MULT>
__global__ __launch_bounds__(256) void pool_kernel(
    const float* __restrict__ score, const float* __restrict__ X,
    const float* __restrict__ mult, float* __restrict__ pooled) {
  const int b = blockIdx.x >> 3;
  const int h = blockIdx.x & 7;
  __shared__ float w[SEQ];
  __shared__ float red[256];
  const int t = threadIdx.x;

  float lmax = -INFINITY;
  for (int s = t; s < SEQ; s += 256) {
    float v = score[((size_t)b * SEQ + s) * NHEAD + h];
    w[s] = v;
    lmax = fmaxf(lmax, v);
  }
  red[t] = lmax;
  __syncthreads();
  for (int off = 128; off > 0; off >>= 1) {
    if (t < off) red[t] = fmaxf(red[t], red[t + off]);
    __syncthreads();
  }
  const float mx = red[0];
  __syncthreads();

  float lsum = 0.f;
  for (int s = t; s < SEQ; s += 256) {
    float e = __expf(w[s] - mx);
    w[s] = e;
    lsum += e;
  }
  red[t] = lsum;
  __syncthreads();
  for (int off = 128; off > 0; off >>= 1) {
    if (t < off) red[t] += red[t + off];
    __syncthreads();
  }
  const float inv = 1.0f / red[0];
  __syncthreads();

  const int d   = t & 63;
  const int grp = t >> 6;  // 0..3
  float acc = 0.f;
  for (int s = grp; s < SEQ; s += 4)
    acc += w[s] * X[((size_t)b * SEQ + s) * HD + h * DHEAD + d];
  red[t] = acc;
  __syncthreads();
  if (grp == 0) {
    float v = (red[d] + red[64 + d] + red[128 + d] + red[192 + d]) * inv;
    if constexpr (HAS_MULT) v *= mult[(size_t)b * HD + h * DHEAD + d];
    pooled[(size_t)b * HD + h * DHEAD + d] = v;
  }
}

// ---------------------------------------------------------------------------
__global__ void cvt_bf16_kernel(const float* __restrict__ s,
                                unsigned short* __restrict__ d, int n) {
  int i = blockIdx.x * 256 + threadIdx.x;
  if (i < n) ((bf16*)d)[i] = (bf16)s[i];
}

// ---------------------------------------------------------------------------
extern "C" void kernel_launch(void* const* d_in, const int* in_sizes, int n_in,
                              void* d_out, int out_size, void* d_ws, size_t ws_size,
                              hipStream_t stream) {
  (void)in_sizes; (void)n_in; (void)out_size; (void)ws_size;
  const float* hs  = (const float*)d_in[0];
  // d_in[1]: attention_mask (unused by reference forward)
  const float* Wq  = (const float*)d_in[2];
  const float* bq  = (const float*)d_in[3];
  const float* Wk  = (const float*)d_in[4];
  const float* bk  = (const float*)d_in[5];
  const float* Wqa = (const float*)d_in[6];
  const float* bqa = (const float*)d_in[7];
  const float* Wka = (const float*)d_in[8];
  const float* bka = (const float*)d_in[9];
  const float* Wt  = (const float*)d_in[10];
  const float* bt  = (const float*)d_in[11];
  float* out = (float*)d_out;

  const size_t M = (size_t)BATCH * SEQ;          // 32768 rows
  char* p = (char*)d_ws;
  float* mq = (float*)p;            p += M * HD * sizeof(float);       // 64 MB
  float* mk = (float*)p;            p += M * HD * sizeof(float);       // 64 MB
  unsigned short* Wqb = (unsigned short*)p; p += (size_t)HD * HD * 2;  // 512 KB
  unsigned short* Wkb = (unsigned short*)p; p += (size_t)HD * HD * 2;
  unsigned short* Wtb = (unsigned short*)p; p += (size_t)HD * HD * 2;
  float* qscore = (float*)p;        p += M * NHEAD * sizeof(float);    // 1 MB
  float* kscore = (float*)p;        p += M * NHEAD * sizeof(float);
  float* pooled_q = (float*)p;      p += (size_t)BATCH * HD * sizeof(float);
  float* pooled_k = (float*)p;      p += (size_t)BATCH * HD * sizeof(float);

  const int nW = HD * HD;
  cvt_bf16_kernel<<<(nW + 255) / 256, 256, 0, stream>>>(Wq, Wqb, nW);
  cvt_bf16_kernel<<<(nW + 255) / 256, 256, 0, stream>>>(Wk, Wkb, nW);
  cvt_bf16_kernel<<<(nW + 255) / 256, 256, 0, stream>>>(Wt, Wtb, nW);

  dim3 gg((unsigned)(M / 128), HD / 128);  // (256, 4)
  // mq = hs @ Wq^T + bq ; mk = hs @ Wk^T + bk
  gemm_bf16_kernel<false, false><<<gg, 256, 0, stream>>>(hs, Wqb, bq, nullptr, nullptr, mq);
  gemm_bf16_kernel<false, false><<<gg, 256, 0, stream>>>(hs, Wkb, bk, nullptr, nullptr, mk);

  // q scores, softmax-pool over S
  score_kernel<false><<<(unsigned)(M / 8), 256, 0, stream>>>(mq, nullptr, Wqa, bqa, qscore);
  pool_kernel<false><<<BATCH * NHEAD, 256, 0, stream>>>(qscore, mq, nullptr, pooled_q);

  // k scores on gated keys (gate fused, mqk never materialized); pooled_k =
  // pooled_q ⊙ pool(mk, k_w)
  score_kernel<true><<<(unsigned)(M / 8), 256, 0, stream>>>(mk, pooled_q, Wka, bka, kscore);
  pool_kernel<true><<<BATCH * NHEAD, 256, 0, stream>>>(kscore, mk, pooled_q, pooled_k);

  // out = (pooled_k ⊙ mq) @ Wt^T + bt + mq   (gate + residual fused in GEMM)
  gemm_bf16_kernel<true, true><<<gg, 256, 0, stream>>>(mq, Wtb, bt, pooled_k, mq, out);
}
